// HFGemma4AttentionAdapter_74792560493257
// MI455X (gfx1250) — matrix operand order repaired and checked by reading, not compiled, not run on hardware
//
#include <hip/hip_runtime.h>
#include <hip/hip_bf16.h>

// ---------------------------------------------------------------------------
// Gemma sliding-window attention for MI455X (gfx1250, wave32, WMMA).
// All matmuls use v_wmma_f32_16x16x32_bf16 (bf16 in, f32 accumulate).
// ---------------------------------------------------------------------------

#define H_Q    8
#define KVH_N  4
#define DH     256
#define SW_WIN 1024
#define S_LEN  2048
#define D_MOD  2048
#define B_N    2

typedef __attribute__((ext_vector_type(16))) __bf16 bf16x16;
typedef __attribute__((ext_vector_type(8)))  float  f32x8;

union FragAB { bf16x16 v; unsigned int u[8]; };

__device__ __forceinline__ unsigned short f2bf(float f) {
  unsigned int u = __float_as_uint(f);
  u += 0x7FFFu + ((u >> 16) & 1u);          // round-to-nearest-even
  return (unsigned short)(u >> 16);
}
__device__ __forceinline__ unsigned int pack2(float a, float b) {
  return (unsigned int)f2bf(a) | ((unsigned int)f2bf(b) << 16);
}

// A operand: 16x32 bf16 tile, row-major in LDS (stride lda elems).
// Lane layout (ISA 7.12.2): row = lane&15; lanes<16 K in {0..7,16..23},
// lanes>=16 K in {8..15,24..31}; each VGPR packs 2 consecutive K.
__device__ __forceinline__ void load_a_frag(const unsigned short* base, int lda,
                                            int lane, FragAB& fr) {
  const int m  = lane & 15;
  const int kb = (lane >> 4) << 3;
  const unsigned short* row = base + m * lda;
#pragma unroll
  for (int i = 0; i < 4; ++i) {
    fr.u[i]     = *(const unsigned int*)(row + kb + 2 * i);
    fr.u[4 + i] = *(const unsigned int*)(row + 16 + kb + 2 * i);
  }
}

// B operand: 32x16 bf16 (KxN). LDS tile stored as [n][k] (contiguous K per
// column). Lane layout: col = lane&15; lanes<16 K in {0..7,16..23},
// lanes>=16 K in {8..15,24..31}.
__device__ __forceinline__ void load_b_frag(const unsigned short* base, int ldb,
                                            int lane, FragAB& fr) {
  const int n = lane & 15;
  const unsigned short* row = base + n * ldb + ((lane >> 4) << 3);
#pragma unroll
  for (int i = 0; i < 4; ++i) {
    fr.u[i]     = *(const unsigned int*)(row + 2 * i);
    fr.u[4 + i] = *(const unsigned int*)(row + 16 + 2 * i);
  }
}

// ---------------------------------------------------------------------------
// C[M,N] = A[M,K] x W[N,K]^T.  A is f32 (converted on load) or bf16.
// Block: 256 threads = 8 waves; tile 128x128, K-step 32.
// Wave w computes rows [16w,16w+16) x all 128 cols -> 8 WMMA / k-step.
// ---------------------------------------------------------------------------
template <bool A_BF16>
__global__ void gemm_nt(const void* Ap, const float* W, float* C,
                        int M, int N, int K) {
  __shared__ __align__(16) unsigned short As[128][32];   // 8 KB
  __shared__ __align__(16) unsigned short Ws[128][32];   // 8 KB
  const int tid  = threadIdx.x;
  const int lane = tid & 31;
  const int wv   = tid >> 5;
  const int gm0  = blockIdx.y * 128;
  const int gn0  = blockIdx.x * 128;
  (void)M;

  f32x8 acc[8];
#pragma unroll
  for (int n = 0; n < 8; ++n)
#pragma unroll
    for (int i = 0; i < 8; ++i) acc[n][i] = 0.0f;

  const float*          Af = (const float*)Ap;
  const unsigned short* Ab = (const unsigned short*)Ap;

  for (int k0 = 0; k0 < K; k0 += 32) {
    // stage tiles: 128 rows x 32 cols each, vectorized 4-elem loads
#pragma unroll
    for (int i = 0; i < 4; ++i) {
      const int slot = i * 256 + tid;          // 0..1023 float4 slots
      const int r = slot >> 3, c4 = (slot & 7) << 2;
      if (A_BF16) {
        const uint2 av =
            *(const uint2*)(Ab + (size_t)(gm0 + r) * K + k0 + c4);
        *(uint2*)(&As[r][c4]) = av;
      } else {
        const float4 av =
            *(const float4*)(Af + (size_t)(gm0 + r) * K + k0 + c4);
        *(uint2*)(&As[r][c4]) = make_uint2(pack2(av.x, av.y),
                                           pack2(av.z, av.w));
      }
      const float4 wvv =
          *(const float4*)(W + (size_t)(gn0 + r) * K + k0 + c4);
      *(uint2*)(&Ws[r][c4]) = make_uint2(pack2(wvv.x, wvv.y),
                                         pack2(wvv.z, wvv.w));
    }
    if (k0 + 32 < K) {  // prefetch next K-slice (global_prefetch_b8)
      __builtin_prefetch((const void*)(W + (size_t)(gn0 + (tid & 127)) * K +
                                       k0 + 32), 0, 3);
      __builtin_prefetch((const void*)((const char*)Ap +
                                       ((size_t)(gm0 + (tid & 127)) * K + k0 +
                                        32) * (A_BF16 ? 2 : 4)), 0, 3);
    }
    __syncthreads();

    FragAB a;
    load_a_frag(&As[wv * 16][0], 32, lane, a);
#pragma unroll
    for (int n = 0; n < 8; ++n) {
      FragAB bf;
      load_b_frag(&Ws[n * 16][0], 32, lane, bf);
      acc[n] = __builtin_amdgcn_wmma_f32_16x16x32_bf16(
          false, a.v, false, bf.v, (short)0, acc[n], false, false);
    }
    __syncthreads();
  }

  const int g = lane >> 4, cn = lane & 15;
#pragma unroll
  for (int n = 0; n < 8; ++n)
#pragma unroll
    for (int r = 0; r < 8; ++r) {
      const int row = gm0 + wv * 16 + r + 8 * g;
      const int col = gn0 + n * 16 + cn;
      C[(size_t)row * N + col] = acc[n][r];
    }
}

// ---------------------------------------------------------------------------
// Fused RMSNorm (eps=1e-6, scale 1+w) + RoPE (base 10000) over head_dim=256.
// One wave per (b,s,head) row; output bf16 in [B][NH][S][DH] layout.
// ---------------------------------------------------------------------------
__global__ void norm_rope_kernel(const float* X, const float* gw,
                                 unsigned short* Out, int NH) {
  const int lane = threadIdx.x & 31;
  const int wv   = threadIdx.x >> 5;
  const int row  = blockIdx.x * 4 + wv;          // over B*S*NH
  const int b    = row / (S_LEN * NH);
  const int rem  = row % (S_LEN * NH);
  const int s    = rem / NH;
  const int h    = rem % NH;
  const float* xr = X + (size_t)row * DH;

  float x1[4], x2[4];
  float ss = 0.0f;
#pragma unroll
  for (int c = 0; c < 4; ++c) {
    const int i = lane * 4 + c;
    x1[c] = xr[i];
    x2[c] = xr[i + 128];
    ss += x1[c] * x1[c] + x2[c] * x2[c];
  }
#pragma unroll
  for (int m = 1; m < 32; m <<= 1) ss += __shfl_xor(ss, m, 32);
  const float inv = rsqrtf(ss * (1.0f / 256.0f) + 1e-6f);

  unsigned short* orow =
      Out + (size_t)(((size_t)b * NH + h) * S_LEN + s) * DH;
#pragma unroll
  for (int c = 0; c < 4; ++c) {
    const int i = lane * 4 + c;
    const float n1 = x1[c] * inv * (1.0f + gw[i]);
    const float n2 = x2[c] * inv * (1.0f + gw[i + 128]);
    // inv_freq = 10000^(-i/128) = exp(-i * ln(1e4)/128)
    const float ang = (float)s * __expf(-(float)i * (9.2103403720f / 128.0f));
    const float cs = __cosf(ang), sn = __sinf(ang);
    orow[i]       = f2bf(n1 * cs - n2 * sn);
    orow[i + 128] = f2bf(n2 * cs + n1 * sn);
  }
}

// V: [B][S][KVH*DH] f32 -> bf16 [B][KVH][S][DH]
__global__ void v_convert_kernel(const float* V, unsigned short* Vb) {
  const int t = blockIdx.x * blockDim.x + threadIdx.x;   // 0..4194303
  const int d = t & 255;
  const int e = (t >> 8) & 3;
  const int s = (t >> 10) & 2047;
  const int b = t >> 21;
  Vb[(((size_t)b * KVH_N + e) * S_LEN + s) * DH + d] = f2bf(V[t]);
}

// ---------------------------------------------------------------------------
// Flash-attention, sliding window 1024, GQA (q head h -> kv head h/2).
// Block: 128 threads (4 waves). Wave = (pair pr, half hf):
//   pr selects 16 query rows (q0+16*pr ..), hf splits work:
//   - scores: hf picks 16 of the 32 keys (8 WMMA / wave, no redundancy)
//   - PV:     hf picks 128 of the 256 output cols (8 WMMA / wave)
// Per-row max/sum partials are merged across the pair through LDS, so the
// accumulator is only 8 x v8f = 64 VGPRs -> no scratch spills.
// ---------------------------------------------------------------------------
__global__ void attn_kernel(const unsigned short* Qb, const unsigned short* Kb,
                            const unsigned short* Vb, unsigned short* Ob) {
  __shared__ __align__(16) unsigned short Qs[32][DH];      // 16 KB
  __shared__ __align__(16) unsigned short Ks[32][DH];      // 16 KB
  __shared__ __align__(16) unsigned short Vs[DH][32];      // 16 KB (transposed)
  __shared__ __align__(16) unsigned short Ps[2][16][32];   //  2 KB
  __shared__ float Mx[2][2][16];                           // rowmax partials
  __shared__ float Sx[2][2][16];                           // rowsum partials

  const int tid  = threadIdx.x;          // 0..127
  const int lane = tid & 31;
  const int wv   = tid >> 5;             // 0..3
  const int pr   = wv >> 1;              // q-tile pair
  const int hf   = wv & 1;               // key-half / col-half
  const int q0   = blockIdx.x * 32;
  const int h    = blockIdx.y;
  const int b    = blockIdx.z;
  const int kvh  = h >> 1;
  const int g    = lane >> 4;
  const int cn   = lane & 15;

  {  // stage 32 query rows (bf16, contiguous) into LDS
    const unsigned int* src = (const unsigned int*)(
        Qb + (((size_t)b * H_Q + h) * S_LEN + q0) * DH);
    unsigned int* dst = (unsigned int*)&Qs[0][0];
    for (int i = tid; i < 32 * (DH / 2); i += 128) dst[i] = src[i];
  }

  f32x8 acc[8];
#pragma unroll
  for (int t = 0; t < 8; ++t)
#pragma unroll
    for (int i = 0; i < 8; ++i) acc[t][i] = 0.0f;
  float mrow[8], lrow[8];
#pragma unroll
  for (int r = 0; r < 8; ++r) { mrow[r] = -1e30f; lrow[r] = 0.0f; }

  int jstart = q0 - (SW_WIN - 1);
  if (jstart < 0) jstart = 0;
  jstart &= ~31;
  const size_t kvbase = ((size_t)b * KVH_N + kvh) * S_LEN * (size_t)DH;

  for (int j0 = jstart; j0 <= q0 + 31; j0 += 32) {
    {  // K tile straight copy; V tile transposed into [dh][key]
      const unsigned int* src =
          (const unsigned int*)(Kb + kvbase + (size_t)j0 * DH);
      unsigned int* dst = (unsigned int*)&Ks[0][0];
      for (int i = tid; i < 32 * (DH / 2); i += 128) dst[i] = src[i];
      const unsigned short* vsrc = Vb + kvbase + (size_t)j0 * DH;
      for (int i = tid; i < 32 * DH; i += 128) {
        const int k = i >> 8, d = i & 255;
        Vs[d][k] = vsrc[k * DH + d];
      }
    }
    __syncthreads();

    // ---- scores: 16 q rows (pr) x 16 keys (hf), K-dim = 256 ----
    f32x8 s;
#pragma unroll
    for (int i = 0; i < 8; ++i) s[i] = 0.0f;
#pragma unroll
    for (int kc = 0; kc < 8; ++kc) {
      FragAB a, kf;
      load_a_frag(&Qs[pr * 16][kc * 32], DH, lane, a);
      load_b_frag(&Ks[hf * 16][kc * 32], DH, lane, kf);
      s = __builtin_amdgcn_wmma_f32_16x16x32_bf16(
          false, a.v, false, kf.v, (short)0, s, false, false);
    }

    // ---- mask + per-half rowmax partials ----
    float sc[8];
#pragma unroll
    for (int r = 0; r < 8; ++r) {
      const int i_g = q0 + 16 * pr + r + 8 * g;
      const int j   = j0 + 16 * hf + cn;
      const bool ok = (j <= i_g) && (i_g - j < SW_WIN);
      float v = ok ? s[r] * 0.0625f : -1e30f;   // 256^-0.5
      sc[r] = v;
      v = fmaxf(v, __shfl_xor(v, 1, 32));
      v = fmaxf(v, __shfl_xor(v, 2, 32));
      v = fmaxf(v, __shfl_xor(v, 4, 32));
      v = fmaxf(v, __shfl_xor(v, 8, 32));
      if (cn == 0) Mx[pr][hf][r + 8 * g] = v;
    }
    __syncthreads();

    // ---- merge maxes, exponentiate, rowsum partials, stage P ----
    float alpha8[8];
#pragma unroll
    for (int r = 0; r < 8; ++r) {
      const int row = r + 8 * g;
      const float tm = fmaxf(Mx[pr][0][row], Mx[pr][1][row]);
      const float mn = fmaxf(mrow[r], tm);
      alpha8[r] = __expf(mrow[r] - mn);
      mrow[r] = mn;
      const float pv = (sc[r] > -1e29f) ? __expf(sc[r] - mn) : 0.0f;
      float rs = pv;
      rs += __shfl_xor(rs, 1, 32);
      rs += __shfl_xor(rs, 2, 32);
      rs += __shfl_xor(rs, 4, 32);
      rs += __shfl_xor(rs, 8, 32);
      if (cn == 0) Sx[pr][hf][row] = rs;
      Ps[pr][row][16 * hf + cn] = f2bf(pv);
    }
    __syncthreads();

    // ---- l update + rescale, then O += P(16x32) x V(32x128-half) ----
#pragma unroll
    for (int r = 0; r < 8; ++r) {
      const int row = r + 8 * g;
      lrow[r] = lrow[r] * alpha8[r] + Sx[pr][0][row] + Sx[pr][1][row];
#pragma unroll
      for (int t = 0; t < 8; ++t) acc[t][r] *= alpha8[r];
    }
    FragAB pa;
    load_a_frag(&Ps[pr][0][0], 32, lane, pa);
#pragma unroll
    for (int t = 0; t < 8; ++t) {
      FragAB vf;
      load_b_frag(&Vs[128 * hf + t * 16][0], 32, lane, vf);
      acc[t] = __builtin_amdgcn_wmma_f32_16x16x32_bf16(
          false, pa.v, false, vf.v, (short)0, acc[t], false, false);
    }
    __syncthreads();
  }

  // normalize and emit bf16 attention output [B][S][H*DH]
#pragma unroll
  for (int r = 0; r < 8; ++r) {
    const int i_g = q0 + 16 * pr + r + 8 * g;
    const float invl = 1.0f / lrow[r];
#pragma unroll
    for (int t = 0; t < 8; ++t) {
      Ob[((size_t)b * S_LEN + i_g) * (H_Q * DH) + h * DH + 128 * hf +
         t * 16 + cn] = f2bf(acc[t][r] * invl);
    }
  }
}

// ---------------------------------------------------------------------------
extern "C" void kernel_launch(void* const* d_in, const int* in_sizes, int n_in,
                              void* d_out, int out_size, void* d_ws,
                              size_t ws_size, hipStream_t stream) {
  (void)in_sizes; (void)n_in; (void)out_size; (void)ws_size;
  const float* hs = (const float*)d_in[0];
  const float* Wq = (const float*)d_in[1];
  const float* Wk = (const float*)d_in[2];
  const float* Wv = (const float*)d_in[3];
  const float* Wo = (const float*)d_in[4];
  const float* qw = (const float*)d_in[5];
  const float* kw = (const float*)d_in[6];
  float* out = (float*)d_out;

  char* ws = (char*)d_ws;
  float*          Qf = (float*)(ws + (size_t)0);           // 33.5 MB
  float*          Kf = (float*)(ws + (size_t)33554432);    // 16 MB
  float*          Vf = (float*)(ws + (size_t)50331648);    // 16 MB
  unsigned short* Qb = (unsigned short*)(ws + (size_t)67108864);  // 16 MB
  unsigned short* Kb = (unsigned short*)(ws + (size_t)83886080);  //  8 MB
  unsigned short* Vb = (unsigned short*)(ws + (size_t)92274688);  //  8 MB
  unsigned short* Ob = (unsigned short*)(ws + (size_t)0);  // aliases dead Qf

  const int M = B_N * S_LEN;  // 4096

  // 1) QKV projections (f32 in, bf16 WMMA, f32 out)
  gemm_nt<false><<<dim3(2048 / 128, M / 128), dim3(256), 0, stream>>>(
      hs, Wq, Qf, M, 2048, D_MOD);
  gemm_nt<false><<<dim3(1024 / 128, M / 128), dim3(256), 0, stream>>>(
      hs, Wk, Kf, M, 1024, D_MOD);
  gemm_nt<false><<<dim3(1024 / 128, M / 128), dim3(256), 0, stream>>>(
      hs, Wv, Vf, M, 1024, D_MOD);

  // 2) fused RMSNorm + RoPE -> bf16 head-major; V -> bf16 head-major
  norm_rope_kernel<<<dim3(B_N * S_LEN * H_Q / 4), dim3(128), 0, stream>>>(
      Qf, qw, Qb, H_Q);
  norm_rope_kernel<<<dim3(B_N * S_LEN * KVH_N / 4), dim3(128), 0, stream>>>(
      Kf, kw, Kb, KVH_N);
  v_convert_kernel<<<dim3(B_N * S_LEN * KVH_N * DH / 256), dim3(256), 0,
                     stream>>>(Vf, Vb);

  // 3) sliding-window flash attention
  attn_kernel<<<dim3(S_LEN / 32, H_Q, B_N), dim3(128), 0, stream>>>(
      Qb, Kb, Vb, Ob);

  // 4) output projection (bf16 A)
  gemm_nt<true><<<dim3(2048 / 128, M / 128), dim3(256), 0, stream>>>(
      Ob, Wo, out, M, D_MOD, 2048);
}
